// VectorQuantizer_5935644803167
// MI455X (gfx1250) — matrix operand order repaired and checked by reading, not compiled, not run on hardware
//
#include <hip/hip_runtime.h>

// ---------------------------------------------------------------------------
// VQ codebook lookup for MI455X (gfx1250, wave32, WMMA).
//   dist argmin  ->  argmax( x.e - 0.5||e||^2 )  via v_wmma_f32_16x16x32_bf16
//   y = emb[idx] gathered from the ORIGINAL f32 codebook (exact output).
//
// Round 3: software double-buffer for B fragments.  Prefetch fragment kc+1
// (and the next tile's first fragment + halfnorm at kc==7, index wrapped)
// before issuing the WMMA pair on the current fragment, so loads overlap
// the XDL work instead of draining to loadcnt==0 before every pair.
// ---------------------------------------------------------------------------

typedef __attribute__((ext_vector_type(16))) __bf16 v16bf;
typedef __attribute__((ext_vector_type(8)))  float  v8f;
typedef __attribute__((ext_vector_type(4)))  unsigned int u32x4;
typedef __attribute__((ext_vector_type(4)))  float  f32x4;

#define N_CODES 8192
#define CODE_DIM 256
#define N_TOKENS (16 * 4096)

static __device__ __forceinline__ unsigned short f2bf_u(float f) {
    union { float f; unsigned int u; } v; v.f = f;
    unsigned int r = v.u + 0x7FFFu + ((v.u >> 16) & 1u);   // round-to-nearest-even
    return (unsigned short)(r >> 16);
}

// ---------------------------------------------------------------------------
// Kernel 1: codebook prep.  One block per code (256 threads = 256 dims).
// ---------------------------------------------------------------------------
__global__ __launch_bounds__(256) void vq_prep(
    const float* __restrict__ emb,
    unsigned short* __restrict__ embB,
    float* __restrict__ halfnorm,
    float* __restrict__ loss_slot) {
    const int code = blockIdx.x;
    const int t = threadIdx.x;
    const float v = emb[(size_t)code * CODE_DIM + t];
    embB[(size_t)code * CODE_DIM + t] = f2bf_u(v);
    float s = v * v;
    #pragma unroll
    for (int o = 16; o; o >>= 1) s += __shfl_down(s, o, 32);
    __shared__ float ls[8];
    if ((t & 31) == 0) ls[t >> 5] = s;
    __syncthreads();
    if (t == 0) {
        float tot = 0.f;
        #pragma unroll
        for (int i = 0; i < 8; ++i) tot += ls[i];
        halfnorm[code] = 0.5f * tot;
        if (code == 0) *loss_slot = 0.0f;
    }
}

// ---------------------------------------------------------------------------
// Kernel 2: WMMA argmin sweep.
//   Block = 256 threads = 8 waves; each wave owns 32 tokens (two 16-row A
//   sets held in registers) and sweeps all 8192 codes, with B fragments
//   software-pipelined one step ahead.
// ---------------------------------------------------------------------------
__global__ __launch_bounds__(256) void vq_argmin(
    const float* __restrict__ x,
    const unsigned short* __restrict__ embB,
    const float* __restrict__ halfnorm,
    int* __restrict__ idx_ws) {
    const int lane = threadIdx.x & 31;
    const int wave = threadIdx.x >> 5;
    const int row  = (blockIdx.x * 8 + wave) * 32;   // first token of this wave
    const int m    = lane & 15;
    const int half = lane >> 4;

    // ---- Load A fragments: 2 sets x 16 tokens x 256 dims, f32 -> bf16.
    // A 16x32 layout (matches B): lane(m,half) holds K = [half*16,+16)
    // of row m, packed 2/VGPR in order.
    v16bf afrag[2][8];
    #pragma unroll
    for (int s = 0; s < 2; ++s) {
        const float* xrow = x + (size_t)(row + s * 16 + m) * CODE_DIM;
        #pragma unroll
        for (int kc = 0; kc < 8; ++kc) {
            const f32x4* p0 = (const f32x4*)(xrow + kc * 32 + half * 16);
            const f32x4* p1 = (const f32x4*)(xrow + kc * 32 + half * 16 + 8);
            f32x4 f0 = p0[0], f1 = p0[1], f2 = p1[0], f3 = p1[1];
            union { unsigned short u[16]; v16bf v; } a;
            a.u[0]  = f2bf_u(f0.x); a.u[1]  = f2bf_u(f0.y);
            a.u[2]  = f2bf_u(f0.z); a.u[3]  = f2bf_u(f0.w);
            a.u[4]  = f2bf_u(f1.x); a.u[5]  = f2bf_u(f1.y);
            a.u[6]  = f2bf_u(f1.z); a.u[7]  = f2bf_u(f1.w);
            a.u[8]  = f2bf_u(f2.x); a.u[9]  = f2bf_u(f2.y);
            a.u[10] = f2bf_u(f2.z); a.u[11] = f2bf_u(f2.w);
            a.u[12] = f2bf_u(f3.x); a.u[13] = f2bf_u(f3.y);
            a.u[14] = f2bf_u(f3.z); a.u[15] = f2bf_u(f3.w);
            afrag[s][kc] = a.v;
        }
    }

    float bestS[2][8];
    int   bestI[2][8];
    #pragma unroll
    for (int s = 0; s < 2; ++s)
        #pragma unroll
        for (int r = 0; r < 8; ++r) { bestS[s][r] = -3.4e38f; bestI[s][r] = 0; }

    // ---- Prime the software pipeline: halfnorm + first B fragment of tile 0.
    float hn = halfnorm[m];
    {
        const u32x4* bp0 = (const u32x4*)(embB + (size_t)m * CODE_DIM + half * 16);
        // q0/q1 hold the prefetched fragment for the *next* pipeline step.
        // (declared below, loaded here via init)
    }
    u32x4 q0 = ((const u32x4*)(embB + (size_t)m * CODE_DIM + half * 16))[0];
    u32x4 q1 = ((const u32x4*)(embB + (size_t)m * CODE_DIM + half * 16))[1];

    // ---- Sweep codebook in 16-code tiles.
    for (int nb = 0; nb < N_CODES; nb += 16) {
        const int n = nb + m;                        // this lane's candidate code
        v8f c0, c1;
        #pragma unroll
        for (int r = 0; r < 8; ++r) { c0[r] = -hn; c1[r] = -hn; }  // fold bias

        // B 32x16 bf16 layout: lane(m,half) holds K = [half*16,+16) of col m.
        const u32x4* bp = (const u32x4*)(embB + (size_t)n * CODE_DIM + half * 16);
        float hn_next = 0.f;
        #pragma unroll
        for (int kc = 0; kc < 8; ++kc) {
            union { u32x4 q[2]; v16bf v; } b;
            b.q[0] = q0;                             // consume prefetched fragment
            b.q[1] = q1;
            if (kc < 7) {
                q0 = bp[(kc + 1) * 4 + 0];           // prefetch next K-chunk
                q1 = bp[(kc + 1) * 4 + 1];
            } else {
                // Prefetch first fragment + halfnorm of the next tile.
                // Wrap the index so the loads are always in-bounds (uniform).
                const int nn = ((nb + 16) & (N_CODES - 1)) + m;
                hn_next = halfnorm[nn];
                const u32x4* bpn =
                    (const u32x4*)(embB + (size_t)nn * CODE_DIM + half * 16);
                q0 = bpn[0];
                q1 = bpn[1];
            }
            // Two independent accumulation chains share one B fragment.
            c0 = __builtin_amdgcn_wmma_f32_16x16x32_bf16(
                     false, afrag[0][kc], false, b.v, (short)0, c0, false, false);
            c1 = __builtin_amdgcn_wmma_f32_16x16x32_bf16(
                     false, afrag[1][kc], false, b.v, (short)0, c1, false, false);
        }
        hn = hn_next;
        // C/D layout: VGPR r: lanes 0-15 -> (M=r, N=lane); 16-31 -> (M=8+r).
        #pragma unroll
        for (int r = 0; r < 8; ++r) {
            if (c0[r] > bestS[0][r]) { bestS[0][r] = c0[r]; bestI[0][r] = n; }
            if (c1[r] > bestS[1][r]) { bestS[1][r] = c1[r]; bestI[1][r] = n; }
        }
    }

    // ---- Argmax reduction across the 16 lanes of each half-wave.
    #pragma unroll
    for (int s = 0; s < 2; ++s) {
        #pragma unroll
        for (int r = 0; r < 8; ++r) {
            float sc = bestS[s][r];
            int   i  = bestI[s][r];
            #pragma unroll
            for (int o = 8; o; o >>= 1) {            // masks 8..1 stay in half
                float os = __shfl_xor(sc, o, 32);
                int   oi = __shfl_xor(i, o, 32);
                if (os > sc || (os == sc && oi < i)) { sc = os; i = oi; }
            }
            if (m == 0) idx_ws[row + s * 16 + half * 8 + r] = i;
        }
    }
}

// ---------------------------------------------------------------------------
// Kernel 3: gather y = emb[idx] (exact f32) + fused MSE loss + idx output.
// ---------------------------------------------------------------------------
__global__ __launch_bounds__(256) void vq_gather(
    const float* __restrict__ x,
    const float* __restrict__ emb,
    const int* __restrict__ idx_ws,
    float* __restrict__ y,
    float* __restrict__ idx_out,
    float* __restrict__ loss,
    float scale) {
    const int token = blockIdx.x;
    const int d = threadIdx.x;
    const int id = idx_ws[token];
    const float e  = emb[(size_t)id * CODE_DIM + d];
    const float xv = x[(size_t)token * CODE_DIM + d];
    y[(size_t)token * CODE_DIM + d] = e;            // straight-through: y == quant
    const float diff = e - xv;
    float s = diff * diff;
    #pragma unroll
    for (int o = 16; o; o >>= 1) s += __shfl_down(s, o, 32);
    __shared__ float ls[8];
    if ((d & 31) == 0) ls[d >> 5] = s;
    __syncthreads();
    if (d == 0) {
        float tot = 0.f;
        #pragma unroll
        for (int i = 0; i < 8; ++i) tot += ls[i];
        atomicAdd(loss, tot * scale);
        idx_out[token] = (float)id;
    }
}

// ---------------------------------------------------------------------------
extern "C" void kernel_launch(void* const* d_in, const int* in_sizes, int n_in,
                              void* d_out, int out_size, void* d_ws, size_t ws_size,
                              hipStream_t stream) {
    const float* x   = (const float*)d_in[0];   // [16,4096,256] f32
    const float* emb = (const float*)d_in[1];   // [8192,256]    f32

    float* y_out    = (float*)d_out;                         // 16*4096*256
    float* idx_out  = y_out + (size_t)N_TOKENS * CODE_DIM;   // 65536
    float* loss_out = idx_out + N_TOKENS;                    // 1

    // Workspace carve-out: bf16 codebook (4MB) + half-norms (32KB) + idx (256KB)
    unsigned short* embB = (unsigned short*)d_ws;
    float* halfnorm = (float*)((char*)d_ws + (size_t)N_CODES * CODE_DIM * 2);
    int*   idx_ws   = (int*)((char*)halfnorm + (size_t)N_CODES * 4);

    vq_prep<<<N_CODES, 256, 0, stream>>>(emb, embB, halfnorm, loss_out);

    // 8 waves/block * 32 tokens/wave = 256 tokens/block -> 256 blocks
    vq_argmin<<<N_TOKENS / 256, 256, 0, stream>>>(x, embB, halfnorm, idx_ws);

    vq_gather<<<N_TOKENS, 256, 0, stream>>>(
        x, emb, idx_ws, y_out, idx_out, loss_out,
        1.0f / ((float)N_TOKENS * (float)CODE_DIM));
}
